// SConvLayer_12524124635879
// MI455X (gfx1250) — matrix operand order repaired and checked by reading, not compiled, not run on hardware
//
#include <hip/hip_runtime.h>
#include <math.h>

typedef __attribute__((ext_vector_type(16))) __bf16 v16bf;
typedef __attribute__((ext_vector_type(8)))  __bf16 v8bf;
typedef __attribute__((ext_vector_type(8)))  float  v8f;
typedef __attribute__((ext_vector_type(4)))  unsigned int v4u;
typedef __attribute__((ext_vector_type(8)))  int v8i;
typedef __attribute__((ext_vector_type(4)))  int v4i;

#define GM 4096
#define GN 1024
#define GK 1024
#define BM 128
#define BN 64
#define BK 32
#define LDST 40   // padded LDS row stride (bf16 elems): 64B data + 16B pad

#define MODE_PLAIN 0
#define MODE_SILU  1
#define MODE_MUL   2

// ---- Tensor Data Mover: 2-D tile (rows x BK bf16) global -> LDS, padded rows ----
__device__ __forceinline__ void tdm_load_tile(
    unsigned ldsOff, const __bf16* gsrc,
    unsigned tensor_d0, unsigned tensor_d1,
    unsigned tile_d0, unsigned tile_d1, unsigned stride0)
{
    unsigned long long ga = (unsigned long long)(uintptr_t)gsrc;
    v4u g0;
    g0[0] = 1u;                                  // count=1 (valid user descriptor)
    g0[1] = ldsOff;                              // lds_addr (bytes)
    g0[2] = (unsigned)ga;                        // global_addr[31:0]
    g0[3] = (unsigned)((ga >> 32) & 0x1FFFFFFu)  // global_addr[56:32]
          | (2u << 30);                          // type = 2 ("image")
    v8i g1;
    g1[0] = (int)((1u << 16)      // data_size = 2 bytes
                | (1u << 20)      // pad_enable
                | (3u << 22)      // pad_interval: 16 DWORDs (= 64B row)
                | (3u << 25));    // pad_amount: 4 DWORDs (= 16B pad)
    g1[1] = (int)((tensor_d0 & 0xFFFFu) << 16);                    // tensor_dim0 lo
    g1[2] = (int)((tensor_d0 >> 16) | ((tensor_d1 & 0xFFFFu) << 16));
    g1[3] = (int)((tensor_d1 >> 16) | (tile_d0 << 16));            // tile_dim0
    g1[4] = (int)tile_d1;                                          // tile_dim1 (tile_dim2=0)
    g1[5] = (int)stride0;                                          // tensor_dim0_stride lo
    g1[6] = 0;                                                     // stride0 hi / stride1 lo
    g1[7] = 0;                                                     // stride1 hi
    v4i g2 = {0, 0, 0, 0};
    v4i g3 = {0, 0, 0, 0};
    v8i g4 = {0, 0, 0, 0, 0, 0, 0, 0};
    __builtin_amdgcn_tensor_load_to_lds(g0, g1, g2, g3, g4, 0);
}

// C[m,n] = sum_k A[m,k] * W[n,k]  (+bias, epilogue). bf16 inputs, WMMA f32 acc.
__global__ __launch_bounds__(256) void gemm_bf16_wmma(
    const __bf16* __restrict__ A, const __bf16* __restrict__ W,
    const float* __restrict__ bias, const float* __restrict__ other,
    float* __restrict__ out, int mode)
{
    __shared__ __bf16 As[2][BM][LDST];
    __shared__ __bf16 Bs[2][BN][LDST];

    const int tid  = threadIdx.x;
    const int m0   = blockIdx.y * BM;
    const int n0   = blockIdx.x * BN;
    const int wave = tid >> 5, lane = tid & 31;
    const int wm   = wave & 3, wn = wave >> 2;     // 4x2 waves of 32x32 sub-tiles

    v8f acc[2][2];
    #pragma unroll
    for (int a = 0; a < 2; ++a)
        #pragma unroll
        for (int b = 0; b < 2; ++b)
            #pragma unroll
            for (int e = 0; e < 8; ++e) acc[a][b][e] = 0.0f;

    const int NK = GK / BK;

    if (wave == 0) {
        tdm_load_tile((unsigned)(uintptr_t)&As[0][0][0], A + (size_t)m0 * GK,
                      GK, GM, BK, BM, GK);
        tdm_load_tile((unsigned)(uintptr_t)&Bs[0][0][0], W + (size_t)n0 * GK,
                      GK, GN, BK, BN, GK);
    }

    for (int it = 0; it < NK; ++it) {
        const int buf = it & 1;
        if (wave == 0) {
            if (it + 1 < NK) {
                const int kt = (it + 1) * BK;
                tdm_load_tile((unsigned)(uintptr_t)&As[buf ^ 1][0][0],
                              A + (size_t)m0 * GK + kt, GK, GM, BK, BM, GK);
                tdm_load_tile((unsigned)(uintptr_t)&Bs[buf ^ 1][0][0],
                              W + (size_t)n0 * GK + kt, GK, GN, BK, BN, GK);
                __builtin_amdgcn_s_wait_tensorcnt((short)2);  // pair `it` done
            } else {
                __builtin_amdgcn_s_wait_tensorcnt((short)0);
            }
        }
        __syncthreads();

        // A 16x32 bf16 frag: lane<16 -> M=lane, elems 0..7=K0..7, 8..15=K16..23
        //                    lane>=16 -> M=lane-16, elems 0..7=K8..15, 8..15=K24..31
        const int hs = lane >> 4;
        v16bf af[2];
        #pragma unroll
        for (int mi = 0; mi < 2; ++mi) {
            const int row = wm * 32 + mi * 16 + (lane & 15);
            v8bf lo = *(const v8bf*)&As[buf][row][hs * 8];
            v8bf hi = *(const v8bf*)&As[buf][row][16 + hs * 8];
            #pragma unroll
            for (int j = 0; j < 8; ++j) { af[mi][j] = lo[j]; af[mi][8 + j] = hi[j]; }
        }
        // B 32x16 frag: lane<16 -> N=lane, elems 0..7=K0..7, 8..15=K16..23 ; lane>=16 -> N=lane-16, elems 0..7=K8..15, 8..15=K24..31
        v16bf bfv[2];
        #pragma unroll
        for (int ni = 0; ni < 2; ++ni) {
            const int col = wn * 32 + ni * 16 + (lane & 15);
            v8bf lo = *(const v8bf*)&Bs[buf][col][hs * 8];
            v8bf hi = *(const v8bf*)&Bs[buf][col][16 + hs * 8];
            #pragma unroll
            for (int j = 0; j < 8; ++j) { bfv[ni][j] = lo[j]; bfv[ni][8 + j] = hi[j]; }
        }

        #pragma unroll
        for (int mi = 0; mi < 2; ++mi)
            #pragma unroll
            for (int ni = 0; ni < 2; ++ni)
                acc[mi][ni] = __builtin_amdgcn_wmma_f32_16x16x32_bf16(
                    false, af[mi], false, bfv[ni], (short)0, acc[mi][ni], false, false);

        __syncthreads();
    }

    // C 16x16 f32 layout: lane L, vgpr r -> (M = r + 8*(L>>4), N = L&15)
    const int nl = lane & 15, mo = (lane >> 4) * 8;
    #pragma unroll
    for (int mi = 0; mi < 2; ++mi) {
        #pragma unroll
        for (int ni = 0; ni < 2; ++ni) {
            const int col = n0 + wn * 32 + ni * 16 + nl;
            const float bcol = bias[col];
            #pragma unroll
            for (int r = 0; r < 8; ++r) {
                const int row = m0 + wm * 32 + mi * 16 + mo + r;
                float v = acc[mi][ni][r] + bcol;
                if (mode == MODE_SILU)      v = v / (1.0f + expf(-v));
                else if (mode == MODE_MUL)  v *= other[(size_t)row * GN + col];
                out[(size_t)row * GN + col] = v;
            }
        }
    }
}

// f32 -> bf16 bulk convert (native cvt path)
__global__ __launch_bounds__(256) void cvt_f32_to_bf16(
    const float* __restrict__ in, __bf16* __restrict__ out, int n)
{
    const int i = (blockIdx.x * 256 + threadIdx.x) * 4;
    if (i + 3 < n) {
        float4 v = *(const float4*)(in + i);
        out[i + 0] = (__bf16)v.x;
        out[i + 1] = (__bf16)v.y;
        out[i + 2] = (__bf16)v.z;
        out[i + 3] = (__bf16)v.w;
    }
}

// ln_da[row,h] = -exp(ln_a[h]) * softplus( x[row,:] . W_dt[h,:] + b_dt[h] )
__global__ __launch_bounds__(256) void dt_kernel(
    const float* __restrict__ x, const float* __restrict__ W_dt,
    const float* __restrict__ b_dt, const float* __restrict__ ln_a,
    float* __restrict__ lnda)
{
    const int row  = blockIdx.x;           // 0..4095
    const int wave = threadIdx.x >> 5;     // head = wave (8 waves)
    const int lane = threadIdx.x & 31;
    const float* xr = x + (size_t)row * 1024;
    const float* wr = W_dt + (size_t)wave * 1024;
    float s = 0.f;
    for (int k = lane; k < 1024; k += 32) s += xr[k] * wr[k];
    #pragma unroll
    for (int m = 16; m > 0; m >>= 1) s += __shfl_xor(s, m, 32);
    if (lane == 0) {
        float v  = s + b_dt[wave];
        float sp = (v > 20.f) ? v : log1pf(expf(v));
        lnda[row * 8 + wave] = -expf(ln_a[wave]) * sp;
    }
}

// Per (b,h): cumsum of ln_da -> e[l]; c0,c1; group-norm moments; hidden_next.
__global__ __launch_bounds__(256) void scan_combine(
    const float* __restrict__ lnda, const float* __restrict__ z,
    const float* __restrict__ hidden, float* __restrict__ dacum,
    float* __restrict__ c0, float* __restrict__ c1,
    float* __restrict__ stats, float* __restrict__ hidden_next)
{
    const int bh = blockIdx.x, b = bh >> 3, h = bh & 7;
    const int t = threadIdx.x;
    __shared__ float sA[256], sB[256];
    __shared__ float eLast;

    float v[8]; float run = 0.f;
    #pragma unroll
    for (int j = 0; j < 8; ++j) {
        const int l = t * 8 + j;
        run += lnda[(size_t)(b * 2048 + l) * 8 + h];
        v[j] = run;
    }
    sA[t] = run; __syncthreads();
    float* cur = sA; float* alt = sB;
    for (int off = 1; off < 256; off <<= 1) {
        float xv = cur[t];
        if (t >= off) xv += cur[t - off];
        alt[t] = xv; __syncthreads();
        float* tmp = cur; cur = alt; alt = tmp;
    }
    const float incl = cur[t];
    const float excl = incl - run;
    #pragma unroll
    for (int j = 0; j < 8; ++j)
        dacum[(size_t)bh * 2048 + t * 8 + j] = expf(excl + v[j]);
    if (t == 255) eLast = expf(incl);
    __syncthreads();

    const int i = t & 127, half = t >> 7;
    float s = 0.f;
    for (int m = h + 1 + half; m < 2048; m += 2)
        s += z[(size_t)(b * 2048 + m) * 1024 + h * 128 + i];
    sA[t] = s; __syncthreads();

    float C0 = 0.f, C1 = 0.f;
    if (t < 128) {
        C1 = hidden[bh * 128 + i] + sA[t] + sA[t + 128];
        C0 = z[(size_t)(b * 2048 + h) * 1024 + h * 128 + i];
        c0[bh * 128 + i] = C0;
        c1[bh * 128 + i] = C1;
        hidden_next[bh * 128 + i] = C0 + eLast * C1;
    }
    __syncthreads();

    for (int q = 0; q < 5; ++q) {
        float val = 0.f;
        if (t < 128)
            val = (q == 0) ? C0 : (q == 1) ? C1 : (q == 2) ? C0 * C0
                : (q == 3) ? C0 * C1 : C1 * C1;
        sA[t] = val; __syncthreads();
        for (int sft = 128; sft > 0; sft >>= 1) {
            if (t < sft) sA[t] += sA[t + sft];
            __syncthreads();
        }
        if (t == 0) stats[bh * 8 + q] = sA[0] * (1.0f / 128.0f);
        __syncthreads();
    }
}

// h_norm (bf16 out) = ((c0 + e*c1) - mean) * rsqrt(var+eps) * gn_w + gn_b
__global__ __launch_bounds__(256) void groupnorm_kernel(
    const float* __restrict__ dacum, const float* __restrict__ c0,
    const float* __restrict__ c1, const float* __restrict__ stats,
    const float* __restrict__ gn_w, const float* __restrict__ gn_b,
    __bf16* __restrict__ hnorm)
{
    const size_t idx = (size_t)blockIdx.x * 256 + threadIdx.x;   // over 4096*1024
    const int row = (int)(idx >> 10), col = (int)(idx & 1023);
    const int b = row >> 11, l = row & 2047, h = col >> 7, i = col & 127;
    const int bh = b * 8 + h;
    const float e  = dacum[(size_t)bh * 2048 + l];
    const float C0 = c0[bh * 128 + i], C1 = c1[bh * 128 + i];
    const float* st = &stats[bh * 8];
    const float mean = st[0] + e * st[1];
    const float ex2  = st[2] + 2.f * e * st[3] + e * e * st[4];
    const float var  = ex2 - mean * mean;
    const float hv   = C0 + e * C1;
    hnorm[idx] = (__bf16)((hv - mean) * rsqrtf(var + 1e-5f) * gn_w[h] + gn_b[h]);
}

extern "C" void kernel_launch(void* const* d_in, const int* in_sizes, int n_in,
                              void* d_out, int out_size, void* d_ws, size_t ws_size,
                              hipStream_t stream) {
    const float* x       = (const float*)d_in[0];
    const float* hidden  = (const float*)d_in[1];
    const float* W_z     = (const float*)d_in[2];
    const float* b_z     = (const float*)d_in[3];
    const float* W_z_act = (const float*)d_in[4];
    const float* b_z_act = (const float*)d_in[5];
    const float* W_y     = (const float*)d_in[6];
    const float* b_y     = (const float*)d_in[7];
    const float* W_y_act = (const float*)d_in[8];
    const float* b_y_act = (const float*)d_in[9];
    const float* W_dt    = (const float*)d_in[10];
    const float* b_dt    = (const float*)d_in[11];
    const float* ln_a    = (const float*)d_in[12];
    const float* gn_w    = (const float*)d_in[13];
    const float* gn_b    = (const float*)d_in[14];

    float* y_out    = (float*)d_out;                   // (2,2048,1024)
    float* hid_next = (float*)d_out + (size_t)GM * GN; // (2,8,128)

    float* ws = (float*)d_ws;
    const size_t PANE = (size_t)GM * GN;               // 4M
    float* za    = ws;                  // silu(x W_z_act^T + b)
    float* z     = ws + PANE;
    float* ya    = ws + 2 * PANE;
    float* lnda  = ws + 3 * PANE;       // 4096*8
    float* dacum = lnda + 32768;        // 16*2048
    float* c0    = dacum + 32768;       // 16*128
    float* c1    = c0 + 2048;
    float* stats = c1 + 2048;           // 16*8
    // bf16 region (starts 16B-aligned)
    __bf16* bfr      = (__bf16*)(stats + 256);
    __bf16* x_bf     = bfr;                    // 4M
    __bf16* wz_bf    = x_bf  + PANE;           // 1M
    __bf16* wza_bf   = wz_bf + (size_t)GN * GK;
    __bf16* wy_bf    = wza_bf + (size_t)GN * GK;
    __bf16* wya_bf   = wy_bf + (size_t)GN * GK;
    __bf16* hnorm_bf = wya_bf + (size_t)GN * GK;  // 4M

    const int WN = GN * GK;            // 1M elems per weight
    cvt_f32_to_bf16<<<dim3(PANE / 1024), dim3(256), 0, stream>>>(x, x_bf, (int)PANE);
    cvt_f32_to_bf16<<<dim3(WN / 1024), dim3(256), 0, stream>>>(W_z,     wz_bf,  WN);
    cvt_f32_to_bf16<<<dim3(WN / 1024), dim3(256), 0, stream>>>(W_z_act, wza_bf, WN);
    cvt_f32_to_bf16<<<dim3(WN / 1024), dim3(256), 0, stream>>>(W_y,     wy_bf,  WN);
    cvt_f32_to_bf16<<<dim3(WN / 1024), dim3(256), 0, stream>>>(W_y_act, wya_bf, WN);

    dim3 gblk(256);
    dim3 ggrd(GN / BN, GM / BM);       // 16 x 32

    gemm_bf16_wmma<<<ggrd, gblk, 0, stream>>>(x_bf, wza_bf, b_z_act, nullptr, za, MODE_SILU);
    gemm_bf16_wmma<<<ggrd, gblk, 0, stream>>>(x_bf, wz_bf,  b_z,     za,      z,  MODE_MUL);
    gemm_bf16_wmma<<<ggrd, gblk, 0, stream>>>(x_bf, wya_bf, b_y_act, nullptr, ya, MODE_SILU);

    dt_kernel<<<dim3(GM), dim3(256), 0, stream>>>(x, W_dt, b_dt, ln_a, lnda);

    scan_combine<<<dim3(16), dim3(256), 0, stream>>>(lnda, z, hidden, dacum, c0, c1,
                                                     stats, hid_next);

    groupnorm_kernel<<<dim3((GM * GN) / 256), dim3(256), 0, stream>>>(
        dacum, c0, c1, stats, gn_w, gn_b, hnorm_bf);

    gemm_bf16_wmma<<<ggrd, gblk, 0, stream>>>(hnorm_bf, wy_bf, b_y, ya, y_out, MODE_MUL);
}
